// CrossAttention_1331439862195
// MI455X (gfx1250) — matrix operand order repaired and checked by reading, not compiled, not run on hardware
//
#include <hip/hip_runtime.h>
#include <hip/hip_bf16.h>

// ---------------------------------------------------------------------------
// CrossAttention for MI455X (gfx1250, wave32, WMMA + async/TDM data movement).
//
// Compute-bound problem: ~69 GFLOP total vs ~50 MB of traffic (fits in the
// 192MB L2; 23.3TB/s makes memory ~2us). So: bf16 WMMA everywhere (f32 acc),
// flash-style fused attention so scores never hit HBM, and tile staging via
// the CDNA5 async/TDM engines where the toolchain exposes them.
//
//   1) cvt f32->bf16 for x_q, x_kv, Wq, Wk, Wv, Wo
//   2) Q = x_q @ Wq^T (2048x1024,K=1024); K/V = x_kv @ Wk/v^T (8192x1024,K=768)
//   3) fused flash attention per (b,h,qtile): _split_heads is a pure reshape,
//      so head (b,h) is contiguous: Q + (b*16+h)*65536 viewed as (1024x64),
//      K/V + (b*16+h)*262144 viewed as (4096x64).
//   4) out = attn @ Wo^T -> f32 into d_out
// ---------------------------------------------------------------------------

typedef unsigned short u16;
typedef __attribute__((ext_vector_type(16))) __bf16 bf16x16;
typedef __attribute__((ext_vector_type(8)))  float  f32x8;

// ---- CDNA5 feature probes --------------------------------------------------
#ifdef __has_builtin
#  if __has_builtin(__builtin_amdgcn_global_load_async_to_lds_b128) && \
      __has_builtin(__builtin_amdgcn_s_wait_asynccnt)
#    define HAVE_ASYNC_LDS 1
#  endif
#  if __has_builtin(__builtin_amdgcn_tensor_load_to_lds) && \
      __has_builtin(__builtin_amdgcn_s_wait_tensorcnt)
#    define HAVE_TDM 1
#  endif
#endif
#ifndef HAVE_ASYNC_LDS
#  define HAVE_ASYNC_LDS 0
#endif
#ifndef HAVE_TDM
#  define HAVE_TDM 0
#endif

#if HAVE_ASYNC_LDS
// Builtin signature (from hipcc diagnostics): (int4 AS1*, int4 AS3*, Ii, Ii).
typedef int v4i_a __attribute__((ext_vector_type(4)));
typedef __attribute__((address_space(1))) void*   as1_void;
typedef __attribute__((address_space(3))) void*   as3_void;
typedef __attribute__((address_space(1))) v4i_a*  as1_v4i;
typedef __attribute__((address_space(3))) v4i_a*  as3_v4i;
#define ASYNC_B128(g, l)                                                      \
  __builtin_amdgcn_global_load_async_to_lds_b128(                             \
      (as1_v4i)(as1_void)(void*)(g), (as3_v4i)(as3_void)(void*)(l), 0, 0)
#define ASYNC_WAIT() __builtin_amdgcn_s_wait_asynccnt(0)
#else
#define ASYNC_B128(g, l) (*(uint4*)(l) = *(const uint4*)(g))
#define ASYNC_WAIT() ((void)0)
#endif

#define WMMA_BF16(A, B, C) \
  __builtin_amdgcn_wmma_f32_16x16x32_bf16(false, (A), false, (B), (short)0, (C), false, false)

__device__ __forceinline__ u16 f2bf(float f) {
  union { float f; unsigned u; } x; x.f = f;
  unsigned r = x.u + 0x7FFFu + ((x.u >> 16) & 1u);   // round-to-nearest-even
  return (u16)(r >> 16);
}
__device__ __forceinline__ f32x8 vzero8() {
  f32x8 z = {0.f, 0.f, 0.f, 0.f, 0.f, 0.f, 0.f, 0.f};
  return z;
}

union Frag { bf16x16 v; uint4 q[2]; };

#if HAVE_TDM
typedef unsigned int v4u __attribute__((ext_vector_type(4)));
typedef int v8i __attribute__((ext_vector_type(8)));
typedef int v4i __attribute__((ext_vector_type(4)));

// 2D TDM load: tile 64 x 64 of 2-byte elements, LDS rows padded +4 DWORDs
// after each 32 DWORDs (-> 144B row stride, matches the 72-half LDS layout).
// D# layout per CDNA5 ISA ch.8 (group0: count/lds_addr/global_addr/type,
// group1: data_size/pad/tensor dims/tile dims/stride).
__device__ __forceinline__ void tdm_load_64x64_bf16(
    const void* gaddr, unsigned lds_byte_addr, unsigned tensor_rows) {
  unsigned long long ga = (unsigned long long)gaddr;
  v4u g0;
  g0.x = 1u;                                            // count=1, user mode
  g0.y = lds_byte_addr;                                 // lds_addr [63:32]
  g0.z = (unsigned)(ga & 0xFFFFFFFFu);                  // global_addr lo
  g0.w = (unsigned)((ga >> 32) & 0x01FFFFFFu) | (2u << 30);  // addr hi | type=2
  const unsigned td0 = 64, tile0 = 64, tile1 = 64, str0 = 64;
  v8i g1;
  g1[0] = (int)((1u << 16) | (1u << 20) | (4u << 22) | (3u << 25));
  //            data=2B     pad_en       pad/32dw     pad_amt=4dw
  g1[1] = (int)(td0 << 16);                             // tensor_dim0 lo
  g1[2] = (int)((td0 >> 16) | (tensor_rows << 16));     // dim0 hi | dim1 lo
  g1[3] = (int)((tensor_rows >> 16) | (tile0 << 16));   // dim1 hi | tile_dim0
  g1[4] = (int)(tile1 & 0xFFFFu);                       // tile_dim1 (tile_dim2=0)
  g1[5] = (int)str0;                                    // tensor_dim0_stride lo
  g1[6] = 0;
  g1[7] = 0;
  v4i zz = {0, 0, 0, 0};
#if defined(__clang_major__) && (__clang_major__ >= 23)
  v8i z8 = {0, 0, 0, 0, 0, 0, 0, 0};
  __builtin_amdgcn_tensor_load_to_lds(g0, g1, zz, zz, z8, 0);  // 6-arg form
#else
  __builtin_amdgcn_tensor_load_to_lds(g0, g1, zz, zz, 0);      // 5-arg form
#endif
}
#endif

// ---------------------------------------------------------------------------
__global__ __launch_bounds__(256) void cvt_f32_to_bf16(
    const float* __restrict__ in, u16* __restrict__ out, int n) {
  int i = blockIdx.x * 256 + threadIdx.x;
  if (i < n) out[i] = f2bf(in[i]);
}

// ---------------------------------------------------------------------------
// C[m,n] = sum_k A[m,k] * W[n,k]   (torch Linear: x @ W^T, W is (out,in))
// Block: 256 threads = 8 waves; C tile 128(M) x 128(N); wave tile 32x64
// (2x4 WMMA accumulators -> 8 WMMA per 12 ds_load_b128 per k-step).
template <int OUT_BF16>
__global__ __launch_bounds__(256) void gemm_nt_bf16(
    const u16* __restrict__ A, const u16* __restrict__ W, void* __restrict__ C,
    int M, int N, int K) {
  __shared__ __align__(16) u16 sA[128 * 40];
  __shared__ __align__(16) u16 sW[128 * 40];

  const int tid = threadIdx.x;
  const int wave = tid >> 5, lane = tid & 31;
  const int laneM = lane & 15, laneH = lane >> 4;
  const int wm = wave >> 1, wn = wave & 1;
  const int m0 = blockIdx.x * 128, n0 = blockIdx.y * 128;

  f32x8 acc[2][4];
#pragma unroll
  for (int i = 0; i < 2; ++i)
#pragma unroll
    for (int j = 0; j < 4; ++j) acc[i][j] = vzero8();

  const int ksteps = K >> 5;
  for (int kt = 0; kt < ksteps; ++kt) {
    __syncthreads();
    // A,W tiles: 128 rows x 32 halfs each = 512 x 16B chunks, 2+2 per thread
#pragma unroll
    for (int i0 = 0; i0 < 2; ++i0) {
      int c = tid + i0 * 256;
      int row = c >> 2, off = c & 3;
      ASYNC_B128(A + (size_t)(m0 + row) * K + kt * 32 + off * 8,
                 &sA[row * 40 + off * 8]);
      ASYNC_B128(W + (size_t)(n0 + row) * K + kt * 32 + off * 8,
                 &sW[row * 40 + off * 8]);
    }
    ASYNC_WAIT();
    __syncthreads();

    Frag af[2], wf[4];
#pragma unroll
    for (int s = 0; s < 2; ++s) {  // A frag: lane half-group K[0..7]&[16..23] / [8..15]&[24..31]
      int row = wm * 32 + s * 16 + laneM;
      int c0 = laneH * 8;
      af[s].q[0] = *(const uint4*)&sA[row * 40 + c0];
      af[s].q[1] = *(const uint4*)&sA[row * 40 + c0 + 16];
    }
#pragma unroll
    for (int s = 0; s < 4; ++s) {  // B frag: lane = one N column, K[0..7]&[16..23] / [8..15]&[24..31]
      int row = wn * 64 + s * 16 + laneM;
      int c0 = laneH * 8;
      wf[s].q[0] = *(const uint4*)&sW[row * 40 + c0];
      wf[s].q[1] = *(const uint4*)&sW[row * 40 + c0 + 16];
    }
#pragma unroll
    for (int i = 0; i < 2; ++i)
#pragma unroll
      for (int j = 0; j < 4; ++j)
        acc[i][j] = WMMA_BF16(af[i].v, wf[j].v, acc[i][j]);
  }

  // Epilogue: C layout -> element e: row = e + 8*laneH, col = laneM
#pragma unroll
  for (int i = 0; i < 2; ++i)
#pragma unroll
    for (int j = 0; j < 4; ++j)
#pragma unroll
      for (int e = 0; e < 8; ++e) {
        int m = m0 + wm * 32 + i * 16 + e + 8 * laneH;
        int n = n0 + wn * 64 + j * 16 + laneM;
        float v = acc[i][j][e];
        if (OUT_BF16) ((u16*)C)[(size_t)m * N + n] = f2bf(v);
        else          ((float*)C)[(size_t)m * N + n] = v;
      }
}

// ---------------------------------------------------------------------------
// Fused flash attention. Grid: (qtile=8, head=16, batch=2), block = 256 = 8 waves.
// Each wave owns 16 queries; K/V streamed in 64-key tiles via TDM/async LDS.
// Scores scaled by 1/head_dim (= 1/64, faithful to reference).
__global__ __launch_bounds__(256) void attn_fused(
    const u16* __restrict__ Q, const u16* __restrict__ K,
    const u16* __restrict__ V, u16* __restrict__ O) {
  constexpr int HD = 64, SQ = 1024, SKV = 4096, H = 16, DQ = 1024;
  __shared__ __align__(16) u16 sK[64 * 72];        // (key, d), stride 72 halfs
  __shared__ __align__(16) u16 sVt[64 * 72];       // transposed: (d, key)
  __shared__ __align__(16) u16 sP[8 * 16 * 72];    // per-wave P (16 x 64)
#if HAVE_TDM
  __shared__ __align__(16) u16 sVraw[64 * 72];     // TDM lands V row-major here
#endif

  const int tid = threadIdx.x, wave = tid >> 5, lane = tid & 31;
  const int laneM = lane & 15, laneH = lane >> 4;
  const int qt = blockIdx.x, h = blockIdx.y, b = blockIdx.z;

  const u16* Qh = Q + (size_t)(b * H + h) * (SQ * HD);
  const u16* Kh = K + (size_t)(b * H + h) * (SKV * HD);
  const u16* Vh = V + (size_t)(b * H + h) * (SKV * HD);
  const int qrow0 = qt * 128 + wave * 16;

  // Q fragments (A-matrix 16x32 layout), kept in registers for all KV tiles
  Frag qa[2];
#pragma unroll
  for (int ks = 0; ks < 2; ++ks) {
    const u16* p = Qh + (size_t)(qrow0 + laneM) * HD + ks * 32 + laneH * 8;
    qa[ks].q[0] = *(const uint4*)p;
    qa[ks].q[1] = *(const uint4*)(p + 16);
  }

  float mrow[8], lrow[8];
  f32x8 accO[4];
#pragma unroll
  for (int j = 0; j < 8; ++j) { mrow[j] = -1e30f; lrow[j] = 0.f; }
#pragma unroll
  for (int n = 0; n < 4; ++n) accO[n] = vzero8();
  const float inv_hd = 1.0f / 64.0f;

  for (int kt = 0; kt < SKV / 64; ++kt) {
    __syncthreads();
#if HAVE_TDM
    // Tensor Data Mover: one wave issues two 64x64 bf16 tile descriptors with
    // LDS padding producing the 144B row stride; TENSORcnt guards completion.
    if (wave == 0) {
      tdm_load_64x64_bf16(Kh + (size_t)kt * 64 * HD,
                          (unsigned)(size_t)(void*)sK, SKV);
      tdm_load_64x64_bf16(Vh + (size_t)kt * 64 * HD,
                          (unsigned)(size_t)(void*)sVraw, SKV);
      __builtin_amdgcn_s_wait_tensorcnt(0);
    }
    __syncthreads();
    // Transpose V (row-major LDS -> (d,key) LDS) for B-fragment contiguity
#pragma unroll
    for (int i0 = 0; i0 < 2; ++i0) {
      int c = tid + i0 * 256;
      int row = c >> 3, off = c & 7;
      union { uint4 q; u16 h[8]; } tv;
      tv.q = *(const uint4*)&sVraw[row * 72 + off * 8];
#pragma unroll
      for (int e = 0; e < 8; ++e) sVt[(off * 8 + e) * 72 + row] = tv.h[e];
    }
#else
    // Stage K via async global->LDS; V needs a transpose so it rides VGPRs.
#pragma unroll
    for (int i0 = 0; i0 < 2; ++i0) {
      int c = tid + i0 * 256;
      int row = c >> 3, off = c & 7;
      ASYNC_B128(Kh + (size_t)(kt * 64 + row) * HD + off * 8,
                 &sK[row * 72 + off * 8]);
      union { uint4 q; u16 h[8]; } tv;
      tv.q = *(const uint4*)(Vh + (size_t)(kt * 64 + row) * HD + off * 8);
#pragma unroll
      for (int e = 0; e < 8; ++e) sVt[(off * 8 + e) * 72 + row] = tv.h[e];
    }
    ASYNC_WAIT();
    __syncthreads();
#endif
    if (kt + 1 < SKV / 64) {  // speculative prefetch of next tile into GL2
      __builtin_prefetch(Kh + (size_t)((kt + 1) * 64 + (tid >> 2)) * HD, 0, 3);
      __builtin_prefetch(Vh + (size_t)((kt + 1) * 64 + (tid >> 2)) * HD, 0, 3);
    }

    // S = Q (16x64) . K^T (64x64keys): 4 N-subtiles x 2 K-chunks of WMMA
    f32x8 accS[4];
#pragma unroll
    for (int n = 0; n < 4; ++n) {
      accS[n] = vzero8();
#pragma unroll
      for (int ks = 0; ks < 2; ++ks) {
        Frag kf;
        int addr = (n * 16 + laneM) * 72 + ks * 32 + laneH * 8;
        kf.q[0] = *(const uint4*)&sK[addr];
        kf.q[1] = *(const uint4*)&sK[addr + 16];
        accS[n] = WMMA_BF16(qa[ks].v, kf.v, accS[n]);
      }
    }

    // Online softmax. C layout: element e of acc -> row e+8*laneH, col laneM.
    float rowm[8];
#pragma unroll
    for (int j = 0; j < 8; ++j)
      rowm[j] = fmaxf(fmaxf(accS[0][j], accS[1][j]), fmaxf(accS[2][j], accS[3][j]));
#pragma unroll
    for (int mask = 1; mask < 16; mask <<= 1)
#pragma unroll
      for (int j = 0; j < 8; ++j)
        rowm[j] = fmaxf(rowm[j], __shfl_xor(rowm[j], mask, 32));

    float corr[8], psum[8];
#pragma unroll
    for (int j = 0; j < 8; ++j) {
      float mn = fmaxf(mrow[j], rowm[j] * inv_hd);
      corr[j] = __expf(mrow[j] - mn);
      mrow[j] = mn;
      psum[j] = 0.f;
    }
    // P = exp(S/hd - m) -> per-wave LDS (C layout -> A layout round trip)
#pragma unroll
    for (int n = 0; n < 4; ++n)
#pragma unroll
      for (int j = 0; j < 8; ++j) {
        float p = __expf(accS[n][j] * inv_hd - mrow[j]);
        psum[j] += p;
        sP[(wave * 16 + j + 8 * laneH) * 72 + n * 16 + laneM] = f2bf(p);
      }
#pragma unroll
    for (int mask = 1; mask < 16; mask <<= 1)
#pragma unroll
      for (int j = 0; j < 8; ++j)
        psum[j] += __shfl_xor(psum[j], mask, 32);
#pragma unroll
    for (int j = 0; j < 8; ++j) lrow[j] = lrow[j] * corr[j] + psum[j];
#pragma unroll
    for (int n = 0; n < 4; ++n)
#pragma unroll
      for (int j = 0; j < 8; ++j) accO[n][j] *= corr[j];

    __syncthreads();  // sP/sVt store->load ordering across the block

    // O += P (16x64keys) . V (64keys x 64d)
    Frag pf[2];
#pragma unroll
    for (int ks = 0; ks < 2; ++ks) {
      int addr = (wave * 16 + laneM) * 72 + ks * 32 + laneH * 8;
      pf[ks].q[0] = *(const uint4*)&sP[addr];
      pf[ks].q[1] = *(const uint4*)&sP[addr + 16];
    }
#pragma unroll
    for (int nd = 0; nd < 4; ++nd)
#pragma unroll
      for (int ks = 0; ks < 2; ++ks) {
        Frag vf;
        int addr = (nd * 16 + laneM) * 72 + ks * 32 + laneH * 8;
        vf.q[0] = *(const uint4*)&sVt[addr];
        vf.q[1] = *(const uint4*)&sVt[addr + 16];
        accO[nd] = WMMA_BF16(pf[ks].v, vf.v, accO[nd]);
      }
  }

  // Epilogue: out[b, q, h*64 + d] = accO / l  (concat-head layout for Wo GEMM)
#pragma unroll
  for (int nd = 0; nd < 4; ++nd)
#pragma unroll
    for (int j = 0; j < 8; ++j) {
      int q = qrow0 + j + 8 * laneH;
      int d = nd * 16 + laneM;
      O[(size_t)(b * SQ + q) * DQ + h * HD + d] = f2bf(accO[nd][j] / lrow[j]);
    }
}

// ---------------------------------------------------------------------------
extern "C" void kernel_launch(void* const* d_in, const int* in_sizes, int n_in,
                              void* d_out, int out_size, void* d_ws, size_t ws_size,
                              hipStream_t stream) {
  const float* x_q  = (const float*)d_in[0];  // (2,1024,1024)
  const float* x_kv = (const float*)d_in[1];  // (2,4096,768)
  const float* Wq   = (const float*)d_in[2];  // (1024,1024)
  const float* Wk   = (const float*)d_in[3];  // (1024,768)
  const float* Wv   = (const float*)d_in[4];  // (1024,768)
  const float* Wo   = (const float*)d_in[5];  // (1024,1024)

  char* ws = (char*)d_ws;                       // ~63 MB used
  u16* xq_bf  = (u16*)(ws + 0);                 //  4 MB
  u16* xkv_bf = (u16*)(ws + 4194304);           // 12 MB
  u16* wq_bf  = (u16*)(ws + 16777216);          //  2 MB
  u16* wk_bf  = (u16*)(ws + 18874368);          //  1.5 MB
  u16* wv_bf  = (u16*)(ws + 20447232);          //  1.5 MB
  u16* wo_bf  = (u16*)(ws + 22020096);          //  2 MB
  u16* Qbf    = (u16*)(ws + 24117248);          //  4 MB
  u16* Kbf    = (u16*)(ws + 28311552);          // 16 MB
  u16* Vbf    = (u16*)(ws + 45088768);          // 16 MB
  u16* Abf    = (u16*)(ws + 61865984);          //  4 MB

  auto cvt = [&](const float* in, u16* out, int n) {
    cvt_f32_to_bf16<<<(n + 255) / 256, 256, 0, stream>>>(in, out, n);
  };
  cvt(x_q,  xq_bf,  2 * 1024 * 1024);
  cvt(x_kv, xkv_bf, 2 * 4096 * 768);
  cvt(Wq,   wq_bf,  1024 * 1024);
  cvt(Wk,   wk_bf,  1024 * 768);
  cvt(Wv,   wv_bf,  1024 * 768);
  cvt(Wo,   wo_bf,  1024 * 1024);

  gemm_nt_bf16<1><<<dim3(16, 8), 256, 0, stream>>>(xq_bf,  wq_bf, Qbf, 2048, 1024, 1024);
  gemm_nt_bf16<1><<<dim3(64, 8), 256, 0, stream>>>(xkv_bf, wk_bf, Kbf, 8192, 1024, 768);
  gemm_nt_bf16<1><<<dim3(64, 8), 256, 0, stream>>>(xkv_bf, wv_bf, Vbf, 8192, 1024, 768);

  attn_fused<<<dim3(8, 16, 2), 256, 0, stream>>>(Qbf, Kbf, Vbf, Abf);

  gemm_nt_bf16<0><<<dim3(16, 8), 256, 0, stream>>>(Abf, wo_bf, d_out, 2048, 1024, 1024);
}
